// Codec_35785667510396
// MI455X (gfx1250) — matrix operand order repaired and checked by reading, not compiled, not run on hardware
//
#include <hip/hip_runtime.h>
#include <math.h>

// ---------------- problem constants ----------------
#define HW           1024
#define NCHROWS      48                   // b*c histogram rows
#define TILES_PER_BC 256                  // 65536 samples / 256 per block
#define NBLOCKS      (48 * TILES_PER_BC)  // 12288
#define RES_F        1048576.0f           // values per histogram row
#define NTOT_D       50331648.0           // total delta count

typedef __attribute__((ext_vector_type(16))) _Float16 v16h;
typedef __attribute__((ext_vector_type(8)))  _Float16 v8h;
typedef __attribute__((ext_vector_type(2)))  _Float16 h2;
typedef __attribute__((ext_vector_type(8)))  float    v8f;

// CDNA5 split-counter wait: order wave-private LDS RAW without a full barrier
__device__ __forceinline__ void lds_fence() {
  asm volatile("s_wait_dscnt 0" ::: "memory");
}

__device__ __forceinline__ int hbin(float v) {
  int idx = (int)floorf((v + 1.0f) * 128.0f);
  return idx < 0 ? 0 : (idx > 255 ? 255 : idx);
}

__global__ void zero_ws_kernel(unsigned* __restrict__ p, int n) {
  int i = blockIdx.x * 256 + threadIdx.x;
  if (i < n) p[i] = 0u;
}

__global__ __launch_bounds__(256)
void codec_main(const float* __restrict__ x,
                const float* __restrict__ W0, const float* __restrict__ b0,
                const float* __restrict__ W1, const float* __restrict__ b1,
                const float* __restrict__ W2, const float* __restrict__ b2,
                const float* __restrict__ W3, const float* __restrict__ b3,
                unsigned* __restrict__ g_hist, float* __restrict__ g_blockSum)
{
  // pre-swizzled augmented B fragments (bias row + const col12 + W3 in col13)
  __shared__ __align__(32) _Float16 s_bf[3 * 16 * 32 * 16];   // 48 KB
  // wave-private staging: two 16-sample tiles per wave
  __shared__ __align__(32) _Float16 s_u[8][2][16][16];   // features (err/x/1.0)
  __shared__ __align__(32) _Float16 s_act[8][2][16][16]; // activations (+pred col13)
  __shared__ float s_pix[8][2][16][16];                  // f32 pixels for residual
  __shared__ unsigned s_h0[256], s_h1[256];
  __shared__ float s_red[256];

  const int tid  = threadIdx.x;
  const int lane = tid & 31;
  const int wv   = tid >> 5;
  const int hsel = lane >> 4;   // half-wave select / tail tile select
  const int mrow = lane & 15;   // A-fragment sample row / tail sample
  const int ncol = lane & 15;   // C/D fragment output column

  const int bc   = blockIdx.x >> 8;     // b*3 + c
  const int tile = blockIdx.x & 255;
  const int cIdx = bc % 3;

  s_h0[tid] = 0u; s_h1[tid] = 0u;

  // ---- pre-swizzle augmented B fragments ----
  {
    const float* W0c = W0 + (size_t)cIdx * 16 * 15 * 12;
    const float* W1c = W1 + (size_t)cIdx * 16 * 144;
    const float* W2c = W2 + (size_t)cIdx * 16 * 144;
    const float* b0c = b0 + (size_t)cIdx * 16 * 12;
    const float* b1c = b1 + (size_t)cIdx * 16 * 12;
    const float* b2c = b2 + (size_t)cIdx * 16 * 12;
    const float* W3c = W3 + (size_t)cIdx * 16 * 12;
    const float* b3c = b3 + (size_t)cIdx * 16;

    for (int idx = tid; idx < 3 * 16 * 32; idx += 256) {
      const int l = idx >> 9;            // layer
      const int r = idx & 511;
      const int k = r >> 5;
      const int n = r & 15;              // B-matrix column (= lane % 16 at use site)
      const int hh = (r >> 4) & 1;
      const int Kd = (l == 0) ? 15 : 12; // weight rows; bias row == Kd
      const float* Wbase = (l == 0) ? (W0c + k * 15 * 12)
                                    : (((l == 1) ? W1c : W2c) + k * 12 * 12);
      const float* bvec = (l == 0) ? (b0c + k * 12)
                                   : ((l == 1) ? (b1c + k * 12) : (b2c + k * 12));
      v16h t;
#pragma unroll
      for (int s = 0; s < 16; ++s) {
        const int K = (s < 8) ? (hh * 8 + s) : (16 + hh * 8 + (s - 8));
        float f = 0.0f;
        if (n < 12) {
          if (K < Kd)        f = Wbase[K * 12 + n];   // weights
          else if (K == Kd)  f = bvec[n];             // bias row (hit by constant-1 feature)
        } else if (n == 12) {
          f = (K == Kd) ? 1.0f : 0.0f;                // propagate 1.0 to next layer
        } else if (n == 13 && l == 2) {               // layer2: fused output layer
          if (K < 12)       f = W3c[k * 12 + K];
          else if (K == 12) f = b3c[k];
        }
        t[s] = (_Float16)f;
      }
      *(v16h*)&s_bf[idx * 16] = t;
    }
  }

  // ---- load 32 consecutive 4x4 blocks per wave (coalesced float4) ----
  const float* xb = x + (size_t)bc * (HW * HW);
  const int s0  = tile * 256 + wv * 32;   // first block of this wave (2 tiles)
  const int by  = s0 >> 8;
  const int bx0 = s0 & 255;
#pragma unroll
  for (int i = 0; i < 4; ++i) {
    const int idx4 = i * 32 + lane;       // 0..127 float4 transfers
    const int tl = idx4 >> 6;             // tile
    const int rr = (idx4 >> 4) & 3;       // row in 4x4 block
    const int m  = idx4 & 15;             // block in tile
    const float4 v = *reinterpret_cast<const float4*>(
        xb + (size_t)(by * 4 + rr) * HW + (size_t)(bx0 + tl * 16 + m) * 4);
    s_pix[wv][tl][m][rr * 4 + 0] = v.x;
    s_pix[wv][tl][m][rr * 4 + 1] = v.y;
    s_pix[wv][tl][m][rr * 4 + 2] = v.z;
    s_pix[wv][tl][m][rr * 4 + 3] = v.w;
    const float vv[4] = {v.x, v.y, v.z, v.w};
#pragma unroll
    for (int j = 0; j < 4; ++j) atomicAdd(&s_h0[hbin(vv[j])], 1u);
  }
  lds_fence();

  // ---- init features: u[m][j] = x[j+1] (j<15), u[15] = 1.0 (bias const) ----
#pragma unroll
  for (int tl = 0; tl < 2; ++tl) {
    v8h t;
#pragma unroll
    for (int j = 0; j < 8; ++j) {
      const int J = hsel * 8 + j;
      float f = (J < 15) ? s_pix[wv][tl][mrow][J + 1] : 1.0f;
      t[j] = (_Float16)f;
    }
    *(v8h*)&s_u[wv][tl][mrow][hsel * 8] = t;
  }
  __syncthreads();   // weights + hist zero + staging all ready

  // per-lane leaky slope: identity on the fused-prediction column 13
  const _Float16 slope = (_Float16)((ncol == 13) ? 1.0f : 0.01f);
  const h2 slope2 = {slope, slope};
  float lsum = 0.0f;

#pragma unroll 1
  for (int k = 0; k < 16; ++k) {
    v16h a0, a1;
    {
      const v8h lo0 = *(const v8h*)&s_u[wv][0][mrow][hsel * 8];
      const v8h lo1 = *(const v8h*)&s_u[wv][1][mrow][hsel * 8];
#pragma unroll
      for (int h = 0; h < 8; ++h) {
        a0[h] = lo0[h]; a0[h + 8] = (_Float16)0.0f;
        a1[h] = lo1[h]; a1[h + 8] = (_Float16)0.0f;
      }
    }

#pragma unroll
    for (int l = 0; l < 3; ++l) {
      if (l > 0) {
        const v8h lo0 = *(const v8h*)&s_act[wv][0][mrow][hsel * 8];
        const v8h lo1 = *(const v8h*)&s_act[wv][1][mrow][hsel * 8];
#pragma unroll
        for (int h = 0; h < 8; ++h) { a0[h] = lo0[h]; a1[h] = lo1[h]; }
      }
      const v16h bf = *(const v16h*)&s_bf[((l * 16 + k) * 32 + lane) * 16];
      v8f acc0 = {};   // inline-0 SRC2 (bias folded into K dim)
      v8f acc1 = {};
      acc0 = __builtin_amdgcn_wmma_f32_16x16x32_f16(false, a0, false, bf,
                                                    (short)0, acc0, false, false);
      acc1 = __builtin_amdgcn_wmma_f32_16x16x32_f16(false, a1, false, bf,
                                                    (short)0, acc1, false, false);
      // packed-f16 leaky: cvt_pk -> pk_mul -> pk_max (identity on col 13)
#pragma unroll
      for (int v = 0; v < 8; v += 2) {
        h2 d0, d1;
        d0[0] = (_Float16)acc0[v]; d0[1] = (_Float16)acc0[v + 1];
        d1[0] = (_Float16)acc1[v]; d1[1] = (_Float16)acc1[v + 1];
        d0 = __builtin_elementwise_max(d0, d0 * slope2);
        d1 = __builtin_elementwise_max(d1, d1 * slope2);
        s_act[wv][0][v + hsel * 8][ncol]     = d0[0];
        s_act[wv][0][v + 1 + hsel * 8][ncol] = d0[1];
        s_act[wv][1][v + hsel * 8][ncol]     = d1[0];
        s_act[wv][1][v + 1 + hsel * 8][ncol] = d1[1];
      }
      lds_fence();
    }

    // ---- residual tail: all 32 lanes (tile = hsel, sample = mrow) ----
    {
      float p = (float)s_act[wv][hsel][mrow][13];   // fused prediction
      p = fminf(1.0f, fmaxf(-1.0f, p));
      float d = s_pix[wv][hsel][mrow][k] - p;
      d = fmodf(d + 1.0f, 2.0f) - 1.0f;
      s_u[wv][hsel][mrow][k] = (_Float16)d;          // feature k for next step
      lsum += d * d;
      atomicAdd(&s_h1[hbin(d)], 1u);
    }
    lds_fence();
  }

  // ---- deterministic sum-of-squares reduction ----
  s_red[tid] = lsum;
  __syncthreads();
  for (int s = 128; s > 0; s >>= 1) {
    if (tid < s) s_red[tid] += s_red[tid + s];
    __syncthreads();
  }
  if (tid == 0) g_blockSum[blockIdx.x] = s_red[0];

  // ---- flush privatized histograms ----
  atomicAdd(&g_hist[(size_t)bc * 256 + tid], s_h0[tid]);
  atomicAdd(&g_hist[(size_t)(NCHROWS + bc) * 256 + tid], s_h1[tid]);
}

__global__ __launch_bounds__(256)
void codec_finalize(const unsigned* __restrict__ g_hist,
                    const float* __restrict__ g_blockSum,
                    float* __restrict__ out)
{
  __shared__ double red[256];
  const int tid = threadIdx.x;

  double e0 = 0.0, e1 = 0.0;
  for (int i = tid; i < NCHROWS * 256; i += 256) {
    const float p0 = (float)g_hist[i] * (1.0f / RES_F);
    if (p0 > 0.0f) e0 -= (double)(p0 * log2f(p0));
    const float p1 = (float)g_hist[NCHROWS * 256 + i] * (1.0f / RES_F);
    if (p1 > 0.0f) e1 -= (double)(p1 * log2f(p1));
  }
  double ss = 0.0;
  for (int i = tid; i < NBLOCKS; i += 256) ss += (double)g_blockSum[i];

  red[tid] = e0; __syncthreads();
  for (int s = 128; s > 0; s >>= 1) { if (tid < s) red[tid] += red[tid + s]; __syncthreads(); }
  const double t0 = red[0]; __syncthreads();

  red[tid] = e1; __syncthreads();
  for (int s = 128; s > 0; s >>= 1) { if (tid < s) red[tid] += red[tid + s]; __syncthreads(); }
  const double t1 = red[0]; __syncthreads();

  red[tid] = ss; __syncthreads();
  for (int s = 128; s > 0; s >>= 1) { if (tid < s) red[tid] += red[tid + s]; __syncthreads(); }
  const double t2 = red[0];

  if (tid == 0) {
    out[0] = 255.0f * sqrtf((float)(t2 / NTOT_D));
    out[1] = (float)(t0 / (8.0 * NCHROWS));
    out[2] = (float)(t1 / (8.0 * NCHROWS));
  }
}

extern "C" void kernel_launch(void* const* d_in, const int* in_sizes, int n_in,
                              void* d_out, int out_size, void* d_ws, size_t ws_size,
                              hipStream_t stream)
{
  const float* x  = (const float*)d_in[0];
  const float* W0 = (const float*)d_in[1];
  const float* b0 = (const float*)d_in[2];
  const float* W1 = (const float*)d_in[3];
  const float* b1 = (const float*)d_in[4];
  const float* W2 = (const float*)d_in[5];
  const float* b2 = (const float*)d_in[6];
  const float* W3 = (const float*)d_in[7];
  const float* b3 = (const float*)d_in[8];

  unsigned* g_hist     = (unsigned*)d_ws;                                 // 2*48*256 u32
  float*    g_blockSum = (float*)((char*)d_ws + (size_t)2 * NCHROWS * 256 * 4);
  float*    out        = (float*)d_out;

  const int histWords = 2 * NCHROWS * 256;
  zero_ws_kernel<<<(histWords + 255) / 256, 256, 0, stream>>>(g_hist, histWords);
  codec_main<<<NBLOCKS, 256, 0, stream>>>(x, W0, b0, W1, b1, W2, b2, W3, b3,
                                          g_hist, g_blockSum);
  codec_finalize<<<1, 256, 0, stream>>>(g_hist, g_blockSum, out);
}
